// DCPNet_82025285419170
// MI455X (gfx1250) — matrix operand order repaired and checked by reading, not compiled, not run on hardware
//
#include <hip/hip_runtime.h>
#include <math.h>

// ---------------------------------------------------------------------------
// Types for CDNA5 WMMA (wave32, 16x16x32 bf16 -> f32)
// ---------------------------------------------------------------------------
typedef __attribute__((ext_vector_type(16))) __bf16 v16bf;
typedef __attribute__((ext_vector_type(8)))  float  v8f;

__device__ inline unsigned short f2bf(float f) {
  // round-to-nearest-even fp32 -> bf16 (bits)
  unsigned int u = __float_as_uint(f);
  unsigned int lsb = (u >> 16) & 1u;
  u += 0x7fffu + lsb;
  return (unsigned short)(u >> 16);
}

union FragBF { v16bf v; unsigned short u[16]; };

// ---------------------------------------------------------------------------
// Implicit-GEMM conv (KSZ=1 or 3) + folded BN + activation epilogue.
//   epi: 0 = none, 1 = relu, 2 = prelu(extra=alpha[Cout]), 3 = add extra + relu
// Block = 256 threads (8 waves) computing a 64(Cout) x 64(pixel) tile:
//   wave w -> M subtile mi = w&3 (16 couts), N group ni = w>>2 (32 pixels,
//   held as two 16x16 f32 accumulators -> 2 chained WMMAs per K-step).
// LDS staging per K-step: As 64x32 bf16 (weights), Bs 64x32 bf16 (patches),
// both k-innermost so per-lane fragment gathers follow ISA 7.12.2:
//   A: lane m=L&15, hi=L>>4; VGPR t<4 -> K=2t+8hi ; t>=4 -> 16+2(t-4)+8hi
//   B: lane n=L&15, hi=L>>4; same K mapping as A
// ---------------------------------------------------------------------------
template<int KSZ>
__global__ __launch_bounds__(256)
void conv_wmma_kernel(const float* __restrict__ x, float* __restrict__ y,
                      const float* __restrict__ w,
                      const float* __restrict__ scale, const float* __restrict__ bias,
                      const float* __restrict__ extra,
                      int B, int Cin, int Cout, int H, int Wd, int epi)
{
  constexpr int PAD = KSZ / 2;
  const int K  = Cin * KSZ * KSZ;
  const int P  = B * H * Wd;
  const int HW = H * Wd;
  const int tid  = threadIdx.x;
  const int wave = tid >> 5;
  const int lane = tid & 31;
  const int mi = wave & 3;              // M subtile (16 couts)
  const int ni = wave >> 2;             // N group (32 pixels)
  const int co0     = blockIdx.y << 6;  // 64 couts / block
  const int pixBase = blockIdx.x << 6;  // 64 pixels / block

  __shared__ unsigned short As[64 * 32];
  __shared__ unsigned short Bs[64 * 32];

  v8f acc0 = {}, acc1 = {};

  // staging coordinates: each thread owns 8 consecutive k of one row
  const int srow = tid >> 2;
  const int skk  = (tid & 3) << 3;

  // decode this thread's staging pixel once
  int pb = 0, pyy = 0, pxx = 0;
  const int spix = pixBase + srow;
  const bool pval = spix < P;
  if (pval) {
    pb = spix / HW; int hw = spix - pb * HW;
    pyy = hw / Wd;  pxx = hw - pyy * Wd;
  }
  const bool cval = (co0 + srow) < Cout;
  const float* wrow = w + (size_t)(co0 + srow) * K + skk;

  for (int k0 = 0; k0 < K; k0 += 32) {
    // ---- stage A (weights, OIHW == [Cout][K] row-major) ----
    #pragma unroll
    for (int i = 0; i < 8; ++i) {
      int k = k0 + skk + i;
      float v = (cval && k < K) ? wrow[k0 + i] : 0.f;
      As[srow * 32 + skk + i] = f2bf(v);
    }
    if (k0 + 32 < K) __builtin_prefetch(wrow + k0 + 32, 0, 1);  // global_prefetch

    // ---- stage B (im2col patches, K-order ci*KSZ*KSZ + kh*KSZ + kw) ----
    #pragma unroll
    for (int i = 0; i < 8; ++i) {
      int k = k0 + skk + i;
      float v = 0.f;
      if (pval && k < K) {
        int ci, kh, kw;
        if (KSZ == 1) { ci = k; kh = 0; kw = 0; }
        else { ci = k / 9; int r = k - ci * 9; kh = r / 3; kw = r - kh * 3; }
        int iy = pyy + kh - PAD, ix = pxx + kw - PAD;
        if (iy >= 0 && iy < H && ix >= 0 && ix < Wd)
          v = x[(((size_t)pb * Cin + ci) * H + iy) * Wd + ix];
      }
      Bs[srow * 32 + skk + i] = f2bf(v);
    }
    __syncthreads();

    // ---- per-lane fragments + 2 chained WMMAs ----
    FragBF fa, fb0, fb1;
    int ml = lane & 15, hi = lane >> 4;
    int ra  = (mi * 16 + ml) * 32;
    int rb0 = (ni * 32 + ml) * 32;
    int rb1 = (ni * 32 + 16 + ml) * 32;
    #pragma unroll
    for (int t = 0; t < 8; ++t) {
      int ka = (t < 4) ? (2 * t + 8 * hi) : (16 + 2 * (t - 4) + 8 * hi);
      fa.u[2 * t]      = As[ra + ka];
      fa.u[2 * t + 1]  = As[ra + ka + 1];
      int kb = ka;
      fb0.u[2 * t]     = Bs[rb0 + kb];
      fb0.u[2 * t + 1] = Bs[rb0 + kb + 1];
      fb1.u[2 * t]     = Bs[rb1 + kb];
      fb1.u[2 * t + 1] = Bs[rb1 + kb + 1];
    }
    acc0 = __builtin_amdgcn_wmma_f32_16x16x32_bf16(false, fa.v, false, fb0.v,
                                                   (short)0, acc0, false, false);
    acc1 = __builtin_amdgcn_wmma_f32_16x16x32_bf16(false, fa.v, false, fb1.v,
                                                   (short)0, acc1, false, false);
    __syncthreads();
  }

  // ---- epilogue: C layout VGPR r -> M = r + 8*hi, N = lane&15 ----
  int nl = lane & 15, hi = lane >> 4;
  #pragma unroll
  for (int a = 0; a < 2; ++a) {
    v8f accv = (a == 0) ? acc0 : acc1;
    int pix = pixBase + ni * 32 + a * 16 + nl;
    if (pix < P) {
      int b  = pix / HW; int hw = pix - b * HW;
      int yy = hw / Wd;  int xx = hw - yy * Wd;
      #pragma unroll
      for (int r = 0; r < 8; ++r) {
        int co = co0 + mi * 16 + r + 8 * hi;
        if (co < Cout) {
          float v = accv[r] * scale[co] + bias[co];
          size_t oidx = (((size_t)b * Cout + co) * H + yy) * Wd + xx;
          if      (epi == 1) v = fmaxf(v, 0.f);
          else if (epi == 2) { float al = extra[co]; v = (v >= 0.f) ? v : al * v; }
          else if (epi == 3) { v += extra[oidx]; v = fmaxf(v, 0.f); }
          y[oidx] = v;
        }
      }
    }
  }
}

// ---------------------------------------------------------------------------
// Fold conv-bias + BN into per-channel scale/bias:  y = scale*conv + bias
// ---------------------------------------------------------------------------
__global__ void fold_bn_kernel(float* __restrict__ scale, float* __restrict__ bias,
                               const float* g, const float* beta,
                               const float* mean, const float* var,
                               const float* cb, int C, int hasBN)
{
  int c = blockIdx.x * blockDim.x + threadIdx.x;
  if (c >= C) return;
  float b0 = cb ? cb[c] : 0.f;
  if (hasBN) {
    float s = g[c] * rsqrtf(var[c] + 1e-5f);
    scale[c] = s;
    bias[c]  = (b0 - mean[c]) * s + beta[c];
  } else {
    scale[c] = 1.f;
    bias[c]  = b0;
  }
}

// ---------------------------------------------------------------------------
// 2x2 block-mean pooling (adaptive avg pool with integer factor 2)
// ---------------------------------------------------------------------------
__global__ void pool2x2_kernel(float* __restrict__ dst, const float* __restrict__ src,
                               int B, int C, int Ho, int Wo)
{
  int idx = blockIdx.x * blockDim.x + threadIdx.x;
  int total = B * C * Ho * Wo;
  if (idx >= total) return;
  int xx = idx % Wo; int t1 = idx / Wo;
  int yy = t1 % Ho;  int t2 = t1 / Ho;
  int c  = t2 % C;   int b  = t2 / C;
  int Hi = 2 * Ho, Wi = 2 * Wo;
  const float* s = src + (((size_t)b * C + c) * Hi + 2 * yy) * Wi + 2 * xx;
  dst[idx] = 0.25f * (s[0] + s[1] + s[Wi] + s[Wi + 1]);
}

// ---------------------------------------------------------------------------
// Channel concat [a ; b] in NCHW
// ---------------------------------------------------------------------------
__global__ void concat_kernel(float* __restrict__ dst, const float* __restrict__ a,
                              const float* __restrict__ bsrc,
                              int B, int Ca, int Cb, int HW)
{
  int idx = blockIdx.x * blockDim.x + threadIdx.x;
  int Ct = Ca + Cb;
  int total = B * Ct * HW;
  if (idx >= total) return;
  int hw = idx % HW; int t1 = idx / HW;
  int c  = t1 % Ct;  int b  = t1 / Ct;
  dst[idx] = (c < Ca) ? a[((size_t)b * Ca + c) * HW + hw]
                      : bsrc[((size_t)b * Cb + (c - Ca)) * HW + hw];
}

// ---------------------------------------------------------------------------
// Fused CARAFE: kernel-prediction 1x1 conv (from f2, CC2 channels) + softmax
// over the 9 taps + neighbor reassembly.
// Streaming form: 36 running dot accumulators, context feature j-loop outer
// (no fv[CC2] array -> low VGPR pressure). kpw rows are uniform per channel
// within a wave -> broadcast loads from L2.
// Faithful to the reference's permute(0,1,5,3,4,2).reshape scramble:
//   out[b,c, x*2+s, t*H+y]   (H == W for all levels here).
// ---------------------------------------------------------------------------
template<int CC2>
__global__ __launch_bounds__(256)
void caun_kernel(float* __restrict__ up, const float* __restrict__ xlow,
                 const float* __restrict__ f2,
                 const float* __restrict__ kpw, const float* __restrict__ kpb,
                 int B, int C, int H, int Wd)
{
  int idx = blockIdx.x * blockDim.x + threadIdx.x;
  int total = B * C * H * Wd;
  if (idx >= total) return;
  int xx = idx % Wd; int t1 = idx / Wd;
  int yy = t1 % H;   int t2 = t1 / H;
  int c  = t2 % C;   int b  = t2 / C;
  const int HW = H * Wd;

  // 36 = 9 taps * 2 * 2 (s,t); channel ch = c*36 + k*4 + s*2 + t
  float m36[36];
  #pragma unroll
  for (int q = 0; q < 36; ++q) m36[q] = kpb[c * 36 + q];

  const float* wbase = kpw + (size_t)(c * 36) * CC2;
  const float* fp    = f2 + (size_t)b * CC2 * HW + yy * Wd + xx;
  for (int j = 0; j < CC2; ++j) {
    float fvj = fp[(size_t)j * HW];
    #pragma unroll
    for (int q = 0; q < 36; ++q)
      m36[q] += wbase[(size_t)q * CC2 + j] * fvj;
  }

  // 9 neighbors with replicate pad
  float nb[9];
  #pragma unroll
  for (int k = 0; k < 9; ++k) {
    int dy = k / 3 - 1, dx = k % 3 - 1;
    int iy = min(max(yy + dy, 0), H - 1);
    int ix = min(max(xx + dx, 0), Wd - 1);
    nb[k] = xlow[(((size_t)b * C + c) * H + iy) * Wd + ix];
  }

  const int H2 = 2 * H, W2 = 2 * Wd;
  for (int s = 0; s < 2; ++s) {
    for (int t = 0; t < 2; ++t) {
      int st = s * 2 + t;
      float mmax = -1e30f;
      #pragma unroll
      for (int k = 0; k < 9; ++k) mmax = fmaxf(mmax, m36[k * 4 + st]);
      float sum = 0.f, v = 0.f;
      #pragma unroll
      for (int k = 0; k < 9; ++k) {
        float e = __expf(m36[k * 4 + st] - mmax);
        sum += e;
        v   += e * nb[k];
      }
      up[(((size_t)b * C + c) * H2 + (xx * 2 + s)) * W2 + (t * H + yy)] = v / sum;
    }
  }
}

// ---------------------------------------------------------------------------
// Normalize memory rows (64 x 32)
// ---------------------------------------------------------------------------
__global__ void norm_mem_kernel(float* __restrict__ mn, const float* __restrict__ mem)
{
  int s = threadIdx.x;
  if (s >= 64) return;
  float ss = 0.f;
  for (int j = 0; j < 32; ++j) { float v = mem[s * 32 + j]; ss += v * v; }
  float inv = 1.f / fmaxf(sqrtf(ss), 1e-12f);
  for (int j = 0; j < 32; ++j) mn[s * 32 + j] = mem[s * 32 + j] * inv;
}

// ---------------------------------------------------------------------------
// Memory attention + residual add: per pixel, cosine attn over 64 slots.
// Online softmax (pass 1: running max/sum; pass 2: recompute dots, weight).
// ---------------------------------------------------------------------------
__global__ __launch_bounds__(256)
void memory_add_kernel(float* __restrict__ out, const float* __restrict__ x,
                       const float* __restrict__ mn, int B, int HW)
{
  __shared__ float sm[64 * 32];
  for (int i = threadIdx.x; i < 64 * 32; i += 256) sm[i] = mn[i];
  __syncthreads();

  int pix = blockIdx.x * 256 + threadIdx.x;
  int total = B * HW;
  if (pix >= total) return;
  int b = pix / HW, hw = pix - b * HW;

  float f[32]; float ss = 0.f;
  #pragma unroll
  for (int j = 0; j < 32; ++j) {
    f[j] = x[((size_t)b * 32 + j) * HW + hw];
    ss += f[j] * f[j];
  }
  float inv = 1.f / fmaxf(sqrtf(ss), 1e-12f);

  // pass 1: online logsumexp over 64 slots
  float mx = -1e30f, sum = 0.f;
  for (int s = 0; s < 64; ++s) {
    float d = 0.f;
    #pragma unroll
    for (int j = 0; j < 32; ++j) d += f[j] * sm[s * 32 + j];
    d *= inv;
    float nm = fmaxf(mx, d);
    sum = sum * __expf(mx - nm) + __expf(d - nm);
    mx = nm;
  }
  float isum = 1.f / sum;

  // pass 2: recompute dots, accumulate reconstruction
  float r[32];
  #pragma unroll
  for (int j = 0; j < 32; ++j) r[j] = 0.f;
  for (int s = 0; s < 64; ++s) {
    float d = 0.f;
    #pragma unroll
    for (int j = 0; j < 32; ++j) d += f[j] * sm[s * 32 + j];
    float a = __expf(d * inv - mx);
    #pragma unroll
    for (int j = 0; j < 32; ++j) r[j] += a * sm[s * 32 + j];
  }
  #pragma unroll
  for (int j = 0; j < 32; ++j)
    out[((size_t)b * 32 + j) * HW + hw] = f[j] + r[j] * isum;
}

// ---------------------------------------------------------------------------
// Host side
// ---------------------------------------------------------------------------
static inline int cdiv(int a, int b) { return (a + b - 1) / b; }

static void launch_conv(hipStream_t st, int ksz, const float* x, float* y, const float* w,
                        const float* sc, const float* bi, const float* extra,
                        int B, int Cin, int Cout, int H, int Wd, int epi)
{
  int P = B * H * Wd;
  dim3 grid(cdiv(P, 64), cdiv(Cout, 64));
  dim3 block(256);
  if (ksz == 3)
    conv_wmma_kernel<3><<<grid, block, 0, st>>>(x, y, w, sc, bi, extra, B, Cin, Cout, H, Wd, epi);
  else
    conv_wmma_kernel<1><<<grid, block, 0, st>>>(x, y, w, sc, bi, extra, B, Cin, Cout, H, Wd, epi);
}

// canonical (dict-insertion) slot -> position in jax-sorted flatten order
static const int kSortedMap[85] = {
  /* x0..x3 */ 81, 82, 83, 84,
  /* caun3  */ 20, 18, 16, 21, 19, 17, 23, 22,
  /* caun2  */ 12, 10,  8, 13, 11,  9, 15, 14,
  /* caun1  */  4,  2,  0,  5,  3,  1,  7,  6,
  /* conv2_1 w1,b1,bn1x4,w2,b2,bn2x4,ws,bs,bnsx4 */
  75, 60, 62, 63, 64, 65, 76, 61, 66, 67, 68, 69, 77, 74, 70, 71, 72, 73,
  /* conv1_2 */
  57, 42, 44, 45, 46, 47, 58, 43, 48, 49, 50, 51, 59, 56, 52, 53, 54, 55,
  /* conv0_3 */
  39, 24, 26, 27, 28, 29, 40, 25, 30, 31, 32, 33, 41, 38, 34, 35, 36, 37,
  /* final_w, final_b, memory */
  79, 78, 80
};

extern "C" void kernel_launch(void* const* d_in, const int* in_sizes, int n_in,
                              void* d_out, int out_size, void* d_ws, size_t ws_size,
                              hipStream_t stream)
{
  const int B = 2;
  // Resolve input ordering (dict-insertion vs jax-sorted pytree flatten).
  bool insertion = (n_in >= 85) && (in_sizes[0] == 2 * 32 * 256 * 256);
  const float* in[85];
  for (int i = 0; i < 85; ++i)
    in[i] = (const float*)d_in[insertion ? i : kSortedMap[i]];

  const float* x0 = in[0];   // (2,32,256,256)
  const float* x1 = in[1];   // (2,64,128,128)
  const float* x2 = in[2];   // (2,128,64,64)
  const float* x3 = in[3];   // (2,256,32,32)
  const float* c3_w1 = in[4];  const float* c3_b1 = in[5];  const float* c3_a1 = in[6];
  const float* c3_w2 = in[7];  const float* c3_b2 = in[8];  const float* c3_a2 = in[9];
  const float* c3_kw = in[10]; const float* c3_kb = in[11];
  const float* c2_w1 = in[12]; const float* c2_b1 = in[13]; const float* c2_a1 = in[14];
  const float* c2_w2 = in[15]; const float* c2_b2 = in[16]; const float* c2_a2 = in[17];
  const float* c2_kw = in[18]; const float* c2_kb = in[19];
  const float* c1_w1 = in[20]; const float* c1_b1 = in[21]; const float* c1_a1 = in[22];
  const float* c1_w2 = in[23]; const float* c1_b2 = in[24]; const float* c1_a2 = in[25];
  const float* c1_kw = in[26]; const float* c1_kb = in[27];
  const float** r2p = &in[28];
  const float** r1p = &in[46];
  const float** r0p = &in[64];
  const float* final_w = in[82];
  const float* final_b = in[83];
  const float* memory  = in[84];

  // ---- workspace bump allocator (floats) ----
  float* W = (float*)d_ws;
  size_t off = 0;
  auto alloc = [&](size_t n) { float* p = W + off; off += n; return p; };

  float* ctx3p = alloc((size_t)B * 128 * 32 * 32);
  float* f1_3  = alloc((size_t)B * 128 * 32 * 32);
  float* f2_3  = alloc((size_t)B * 64  * 32 * 32);
  float* x3u   = alloc((size_t)B * 256 * 64 * 64);
  float* cat2  = alloc((size_t)B * 384 * 64 * 64);
  float* res2  = alloc((size_t)B * 128 * 64 * 64);
  float* t2a   = alloc((size_t)B * 128 * 64 * 64);
  float* x2_1  = alloc((size_t)B * 128 * 64 * 64);
  float* ctx2p = alloc((size_t)B * 64  * 64 * 64);
  float* f1_2  = alloc((size_t)B * 64  * 64 * 64);
  float* f2_2  = alloc((size_t)B * 32  * 64 * 64);
  float* x2u   = alloc((size_t)B * 128 * 128 * 128);
  float* cat1  = alloc((size_t)B * 192 * 128 * 128);
  float* res1  = alloc((size_t)B * 64  * 128 * 128);
  float* t1a   = alloc((size_t)B * 64  * 128 * 128);
  float* x1_2  = alloc((size_t)B * 64  * 128 * 128);
  float* ctx1p = alloc((size_t)B * 32  * 128 * 128);
  float* f1_1  = alloc((size_t)B * 32  * 128 * 128);
  float* f2_1  = alloc((size_t)B * 16  * 128 * 128);
  float* x1u   = alloc((size_t)B * 64  * 256 * 256);
  float* cat0  = alloc((size_t)B * 96  * 256 * 256);
  float* res0  = alloc((size_t)B * 32  * 256 * 256);
  float* t0a   = alloc((size_t)B * 32  * 256 * 256);
  float* x0_3  = alloc((size_t)B * 32  * 256 * 256);
  float* x0m   = alloc((size_t)B * 32  * 256 * 256);
  float* mem_n = alloc(64 * 32);
  (void)ws_size;

  auto fold = [&](const float* g, const float* bb, const float* mm, const float* vv,
                  const float* cb, int C, int hasBN, float** sc, float** bi) {
    *sc = alloc(C); *bi = alloc(C);
    fold_bn_kernel<<<cdiv(C, 64), 64, 0, stream>>>(*sc, *bi, g, bb, mm, vv, cb, C, hasBN);
  };

  auto res_block = [&](const float** p, const float* xin, float* rbuf, float* tbuf,
                       float* yout, int Cin, int Cout, int H, int Wd) {
    // p: w1,b1, bn1{g,bb,m,v}, w2,b2, bn2{..}, ws,bs, bns{..}
    float *sS, *bS, *s1, *b1f, *s2, *b2f;
    fold(p[14], p[15], p[16], p[17], p[13], Cout, 1, &sS, &bS);   // shortcut bns
    fold(p[2],  p[3],  p[4],  p[5],  p[1],  Cout, 1, &s1, &b1f);  // bn1
    fold(p[8],  p[9],  p[10], p[11], p[7],  Cout, 1, &s2, &b2f);  // bn2
    launch_conv(stream, 1, xin, rbuf, p[12], sS, bS, nullptr, B, Cin, Cout, H, Wd, 0);
    launch_conv(stream, 3, xin, tbuf, p[0],  s1, b1f, nullptr, B, Cin, Cout, H, Wd, 1);
    launch_conv(stream, 3, tbuf, yout, p[6], s2, b2f, rbuf,    B, Cout, Cout, H, Wd, 3);
  };

  // ================= caun3: x3 (256ch @32) upsampled with ctx x2 ============
  {
    int np = B * 128 * 32 * 32;
    pool2x2_kernel<<<cdiv(np, 256), 256, 0, stream>>>(ctx3p, x2, B, 128, 32, 32);
    float *s1, *b1f, *s2, *b2f;
    fold(nullptr, nullptr, nullptr, nullptr, c3_b1, 128, 0, &s1, &b1f);
    fold(nullptr, nullptr, nullptr, nullptr, c3_b2, 64,  0, &s2, &b2f);
    launch_conv(stream, 3, ctx3p, f1_3, c3_w1, s1, b1f, c3_a1, B, 128, 128, 32, 32, 2);
    launch_conv(stream, 3, f1_3,  f2_3, c3_w2, s2, b2f, c3_a2, B, 128, 64,  32, 32, 2);
    int nt = B * 256 * 32 * 32;
    caun_kernel<64><<<cdiv(nt, 256), 256, 0, stream>>>(x3u, x3, f2_3, c3_kw, c3_kb, B, 256, 32, 32);
  }
  // ================= conv2_1 ================================================
  {
    int HW = 64 * 64;
    int nt = B * 384 * HW;
    concat_kernel<<<cdiv(nt, 256), 256, 0, stream>>>(cat2, x2, x3u, B, 128, 256, HW);
    res_block(r2p, cat2, res2, t2a, x2_1, 384, 128, 64, 64);
  }
  // ================= caun2 ==================================================
  {
    int np = B * 64 * 64 * 64;
    pool2x2_kernel<<<cdiv(np, 256), 256, 0, stream>>>(ctx2p, x1, B, 64, 64, 64);
    float *s1, *b1f, *s2, *b2f;
    fold(nullptr, nullptr, nullptr, nullptr, c2_b1, 64, 0, &s1, &b1f);
    fold(nullptr, nullptr, nullptr, nullptr, c2_b2, 32, 0, &s2, &b2f);
    launch_conv(stream, 3, ctx2p, f1_2, c2_w1, s1, b1f, c2_a1, B, 64, 64, 64, 64, 2);
    launch_conv(stream, 3, f1_2,  f2_2, c2_w2, s2, b2f, c2_a2, B, 64, 32, 64, 64, 2);
    int nt = B * 128 * 64 * 64;
    caun_kernel<32><<<cdiv(nt, 256), 256, 0, stream>>>(x2u, x2_1, f2_2, c2_kw, c2_kb, B, 128, 64, 64);
  }
  // ================= conv1_2 ================================================
  {
    int HW = 128 * 128;
    int nt = B * 192 * HW;
    concat_kernel<<<cdiv(nt, 256), 256, 0, stream>>>(cat1, x1, x2u, B, 64, 128, HW);
    res_block(r1p, cat1, res1, t1a, x1_2, 192, 64, 128, 128);
  }
  // ================= caun1 ==================================================
  {
    int np = B * 32 * 128 * 128;
    pool2x2_kernel<<<cdiv(np, 256), 256, 0, stream>>>(ctx1p, x0, B, 32, 128, 128);
    float *s1, *b1f, *s2, *b2f;
    fold(nullptr, nullptr, nullptr, nullptr, c1_b1, 32, 0, &s1, &b1f);
    fold(nullptr, nullptr, nullptr, nullptr, c1_b2, 16, 0, &s2, &b2f);
    launch_conv(stream, 3, ctx1p, f1_1, c1_w1, s1, b1f, c1_a1, B, 32, 32, 128, 128, 2);
    launch_conv(stream, 3, f1_1,  f2_1, c1_w2, s2, b2f, c1_a2, B, 32, 16, 128, 128, 2);
    int nt = B * 64 * 128 * 128;
    caun_kernel<16><<<cdiv(nt, 256), 256, 0, stream>>>(x1u, x1_2, f2_1, c1_kw, c1_kb, B, 64, 128, 128);
  }
  // ================= conv0_3 ================================================
  {
    int HW = 256 * 256;
    int nt = B * 96 * HW;
    concat_kernel<<<cdiv(nt, 256), 256, 0, stream>>>(cat0, x0, x1u, B, 32, 64, HW);
    res_block(r0p, cat0, res0, t0a, x0_3, 96, 32, 256, 256);
  }
  // ================= memory attention + residual ============================
  {
    int HW = 256 * 256;
    norm_mem_kernel<<<1, 64, 0, stream>>>(mem_n, memory);
    memory_add_kernel<<<cdiv(B * HW, 256), 256, 0, stream>>>(x0m, x0_3, mem_n, B, HW);
  }
  // ================= final 1x1 conv -> d_out ================================
  {
    float *sf, *bf;
    fold(nullptr, nullptr, nullptr, nullptr, final_b, 2, 0, &sf, &bf);
    launch_conv(stream, 1, x0m, (float*)d_out, final_w, sf, bf, nullptr, B, 32, 2, 256, 256, 0);
  }
  (void)out_size;
}
